// RSNN_3521873183653
// MI455X (gfx1250) — matrix operand order repaired and checked by reading, not compiled, not run on hardware
//
#include <hip/hip_runtime.h>

// RSNN forward for MI455X (gfx1250, wave32, WMMA).
// Grid: 64 workgroups x 256 threads (8 waves). Each WG owns BM=16 batch rows
// for all T=50 steps (recurrence is row-local -> no grid sync needed).
// GEMMs in f16 WMMA (spikes are exact {0,1}; weights f16-converted once).
// B fragments are explicitly software-pipelined (4-deep rolling queue) so the
// WMMAs never wait on a just-issued load.

#define B_SZ   1024
#define T_WIN  50
#define NIN    700
#define NH     1024
#define NOUT   20
#define BM     16

#define KIH    704      // NIN padded to multiple of 32
#define XST    712      // xA LDS row stride (halfs, mult of 8 for 16B align)
#define SST    1032     // spike LDS row stride (halfs)
#define THRESH 0.3f
#define DECAY  0.3f

typedef _Float16 v4h  __attribute__((ext_vector_type(4)));
typedef _Float16 v8h  __attribute__((ext_vector_type(8)));
typedef _Float16 v16h __attribute__((ext_vector_type(16)));
typedef float    v4f  __attribute__((ext_vector_type(4)));
typedef float    v8f  __attribute__((ext_vector_type(8)));

// workspace layout (bytes) -- regions consecutive on purpose: the pipelined
// GEMM overfetches <=64B past each weight region into the next one.
#define WS_WIH 0
#define WS_WHH (NH * KIH * 2)                  // 1,441,792
#define WS_WHO (WS_WHH + NH * NH * 2)          // 3,538,944
#define WS_END (WS_WHO + 32 * NH * 2)          // 3,604,480

// A-fragment (16x32 f16): lane holds K {kh..kh+7} and {kh+16..kh+23}, kh=8*(lane>=16)
__device__ __forceinline__ v16h load_a(const _Float16* p) {
  v8h lo = *(const v8h*)(p);
  v8h hi = *(const v8h*)(p + 16);
  return __builtin_shufflevector(lo, hi, 0,1,2,3,4,5,6,7,8,9,10,11,12,13,14,15);
}
// B-fragment (32x16 f16): lane holds K {kb..kb+7} and {kb+16..kb+23}, kb=8*(lane>=16)
__device__ __forceinline__ v16h load_b(const _Float16* p) {
  v8h lo = *(const v8h*)(p);
  v8h hi = *(const v8h*)(p + 16);
  return __builtin_shufflevector(lo, hi, 0,1,2,3,4,5,6,7,8,9,10,11,12,13,14,15);
}

// One N=128 GEMM slice: acc[0..7] += A(16xKC*32) * B-slice. B fragments are
// kept in a 4-deep rolling queue: load for step s+4 is issued before the WMMA
// consuming step s, so waits land on loads issued 4 WMMAs earlier.
template <int KC, int KP>
__device__ __forceinline__ void gemm_block(const _Float16* __restrict__ bw,
                                           const _Float16* aLds, int aOff,
                                           v8f (&acc)[8]) {
  v16h bq[4];
  #pragma unroll
  for (int i = 0; i < 4; ++i) bq[i] = load_b(bw + (size_t)i * 16 * KP);
  v16h a = load_a(aLds + aOff);
  #pragma unroll 1
  for (int kc = 0; kc < KC; ++kc) {
    v16h an = load_a(aLds + aOff + (kc + 1) * 32);   // tail overfetch unused
    #pragma unroll
    for (int nt = 0; nt < 8; ++nt) {
      // issue fragment for flattened step (kc*8+nt)+4 (tail overfetch unused)
      v16h bn = load_b(bw + (size_t)((nt + 4) & 7) * 16 * KP +
                       (kc + ((nt + 4) >> 3)) * 32);
      acc[nt] = __builtin_amdgcn_wmma_f32_16x16x32_f16(
          false, a, false, bq[nt & 3], (short)0, acc[nt], false, false);
      bq[nt & 3] = bn;
    }
    a = an;
  }
}

__global__ void rsnn_prep(const float* __restrict__ Wih,
                          const float* __restrict__ Whh,
                          const float* __restrict__ Who,
                          _Float16* __restrict__ wih16,
                          _Float16* __restrict__ whh16,
                          _Float16* __restrict__ who16) {
  const int N1 = NH * KIH, N2 = NH * NH, N3 = 32 * NH;
  const int stride = gridDim.x * blockDim.x;
  for (int i = blockIdx.x * blockDim.x + threadIdx.x; i < N1 + N2 + N3; i += stride) {
    if (i < N1) {
      int j = i / KIH, k = i - j * KIH;
      wih16[i] = (_Float16)(k < NIN ? Wih[j * NIN + k] : 0.0f);
    } else if (i < N1 + N2) {
      int i2 = i - N1;
      whh16[i2] = (_Float16)Whh[i2];
    } else {
      int i3 = i - N1 - N2;
      int j = i3 >> 10, k = i3 & (NH - 1);
      who16[i3] = (_Float16)(j < NOUT ? Who[j * NH + k] : 0.0f);
    }
  }
}

__global__ __launch_bounds__(256) void rsnn_main(
    const float* __restrict__ input,
    const _Float16* __restrict__ wih,
    const _Float16* __restrict__ whh,
    const _Float16* __restrict__ who,
    float* __restrict__ out) {
  __shared__ _Float16 spk[BM * SST];   // hidden spikes, f16 row-major (33,024 B)
  __shared__ _Float16 xA [BM * XST];   // binarized input tile       (22,784 B)
  __shared__ float mem2buf[16 * 32];   // output-layer partial sums   (2,048 B)
  __shared__ float h2mem[320];
  __shared__ float h2sum[320];

  const int tid  = threadIdx.x;
  const int lane = tid & 31;
  const int wv   = tid >> 5;        // wave 0..7, owns N columns [wv*128, wv*128+128)
  const int col  = lane & 15;
  const int hi16 = lane >> 4;
  const int kha  = hi16 * 8;        // A-frag K sub-offset (halfs)
  const int khb  = hi16 * 8;        // B-frag K sub-offset (halfs)
  const int rb   = blockIdx.x * BM;

  for (int i = tid; i < BM * SST; i += 256) spk[i] = (_Float16)0.0f;
  for (int i = tid; i < BM * XST; i += 256) xA[i]  = (_Float16)0.0f;  // keeps K-pad zero
  for (int i = tid; i < 320; i += 256) { h2mem[i] = 0.0f; h2sum[i] = 0.0f; }

  v8f zero8;
  #pragma unroll
  for (int i = 0; i < 8; ++i) zero8[i] = 0.0f;
  v8f memst[8];
  #pragma unroll
  for (int nt = 0; nt < 8; ++nt) memst[nt] = zero8;

  const int aX = col * XST + kha;
  const int aS = col * SST + kha;
  const _Float16* wihW = wih + (size_t)(wv * 128 + col) * KIH + khb;
  const _Float16* whhW = whh + (size_t)(wv * 128 + col) * NH  + khb;
  const _Float16* whoW = who + (size_t)col * NH + khb;

  for (int t = 0; t < T_WIN; ++t) {
    // ---- phase 1: stage x_bin tile (vectorized: each row = 175 aligned float4),
    //               zero output reduction buffer ----
    for (int i = tid; i < BM * 175; i += 256) {
      int r  = i / 175;
      int c4 = i - r * 175;                       // float4 index within row
      const v4f x4 = *(const v4f*)(input + ((size_t)(rb + r) * T_WIN + t) * NIN + c4 * 4);
      v4h h;
      #pragma unroll
      for (int e = 0; e < 4; ++e) h[e] = (_Float16)(x4[e] > 0.0f ? 1.0f : 0.0f);
      *(v4h*)(&xA[r * XST + c4 * 4]) = h;
    }
    for (int i = tid; i < 512; i += 256) mem2buf[i] = 0.0f;
    __syncthreads();

    // prefetch next step's input tile into cache while the GEMMs run
    // (speculative global_prefetch_b8: no LOADcnt, OOB tail silently dropped)
    if (t + 1 < T_WIN) {
      for (int i = tid; i < BM * 22; i += 256) {  // 22 x 128B lines per 2800B row
        int r = i / 22, l = i - r * 22;
        __builtin_prefetch(input + ((size_t)(rb + r) * T_WIN + t + 1) * NIN + l * 32, 0, 1);
      }
    }

    // ---- phase 2: mem = x@Wih^T + spk_{t-1}@Whh^T (f16 WMMA, f32 acc) ----
    v8f acc[8];
    #pragma unroll
    for (int nt = 0; nt < 8; ++nt) acc[nt] = zero8;

    gemm_block<KIH / 32, KIH>(wihW, xA,  aX, acc);   // input GEMM  (K=704)
    gemm_block<NH  / 32, NH >(whhW, spk, aS, acc);   // recurrent GEMM (K=1024)
    __syncthreads();  // all waves done reading spk_{t-1} before overwrite

    // ---- phase 3: neuron update; write spikes (C layout: reg r -> row, col=lane&15)
    // note: h0_mem is zero wherever spk fired, so the (1-spk) factor is redundant.
    #pragma unroll
    for (int nt = 0; nt < 8; ++nt) {
      const int nb = wv * 128 + nt * 16 + col;
      #pragma unroll
      for (int r = 0; r < 8; ++r) {
        float m = acc[nt][r] + memst[nt][r] * DECAY;
        int row = hi16 * 8 + r;
        spk[row * SST + nb] = (_Float16)(m > THRESH ? 1.0f : 0.0f);
        memst[nt][r] = (m < THRESH) ? m : 0.0f;
      }
    }
    __syncthreads();

    // ---- phase 4: output GEMM spk_t @ Who^T; waves split K=1024 into 8 slices
    v8f a2[2];
    a2[0] = zero8; a2[1] = zero8;
    #pragma unroll
    for (int kc = 0; kc < 4; ++kc) {
      const int k0 = (wv * 4 + kc) * 32;
      v16h a = load_a(&spk[aS + k0]);
      #pragma unroll
      for (int nt = 0; nt < 2; ++nt) {
        v16h b = load_b(whoW + (size_t)nt * 16 * NH + k0);
        a2[nt] = __builtin_amdgcn_wmma_f32_16x16x32_f16(
            false, a, false, b, (short)0, a2[nt], false, false);
      }
    }
    #pragma unroll
    for (int nt = 0; nt < 2; ++nt) {
      #pragma unroll
      for (int r = 0; r < 8; ++r) {
        int row = hi16 * 8 + r;
        atomicAdd(&mem2buf[row * 32 + nt * 16 + col], a2[nt][r]);  // ds_add_f32
      }
    }
    __syncthreads();

    // ---- phase 5: output neuron update (16x20 state in LDS) ----
    for (int idx = tid; idx < 320; idx += 256) {
      int r = idx & 15, o = idx >> 4;
      float m2 = h2mem[idx] * DECAY + mem2buf[r * 32 + o];
      h2sum[idx] += (m2 > THRESH) ? 1.0f : 0.0f;
      h2mem[idx]  = (m2 < THRESH) ? m2 : 0.0f;
    }
    __syncthreads();
  }

  for (int idx = tid; idx < 320; idx += 256) {
    int r = idx & 15, o = idx >> 4;
    out[(size_t)(rb + r) * NOUT + o] = h2sum[idx] * (1.0f / T_WIN);
  }
}

extern "C" void kernel_launch(void* const* d_in, const int* in_sizes, int n_in,
                              void* d_out, int out_size, void* d_ws, size_t ws_size,
                              hipStream_t stream) {
  (void)in_sizes; (void)n_in; (void)out_size; (void)ws_size;  // needs >= 3.7 MB ws
  const float* input = (const float*)d_in[0];
  const float* Wih   = (const float*)d_in[1];
  const float* Whh   = (const float*)d_in[2];
  const float* Who   = (const float*)d_in[3];
  char* ws = (char*)d_ws;
  _Float16* wih16 = (_Float16*)(ws + WS_WIH);
  _Float16* whh16 = (_Float16*)(ws + WS_WHH);
  _Float16* who16 = (_Float16*)(ws + WS_WHO);

  rsnn_prep<<<512, 256, 0, stream>>>(Wih, Whh, Who, wih16, whh16, who16);
  rsnn_main<<<B_SZ / BM, 256, 0, stream>>>(input, wih16, whh16, who16, (float*)d_out);
}
